// MultiHeadedAttention_55697135895191
// MI455X (gfx1250) — matrix operand order repaired and checked by reading, not compiled, not run on hardware
//
#include <hip/hip_runtime.h>
#include <hip/hip_bf16.h>

// Problem constants (match the reference)
#define BB   4
#define SS   1024
#define EE   512
#define HH   8
#define DKK  64

typedef unsigned short u16;
typedef __attribute__((ext_vector_type(16))) __bf16 bf16x16;
typedef __attribute__((ext_vector_type(8)))  float  f32x8;
typedef int v4i __attribute__((vector_size(16)));   // matches builtin's V4i

// 32-byte WMMA operand: 16 bf16 loaded as two 16B chunks
union OpPack { bf16x16 v; uint4 q[2]; u16 u[16]; };

#if defined(__HIP_DEVICE_COMPILE__) && __has_builtin(__builtin_amdgcn_global_load_async_to_lds_b128)
#define ASYNC_LDS 1
#else
#define ASYNC_LDS 0
#endif

__device__ __forceinline__ void lds_copy16B(const u16* g, u16* l) {
#if ASYNC_LDS
  // GLOBAL_LOAD_ASYNC_TO_LDS_B128: memory -> LDS without a VGPR round trip,
  // tracked with ASYNCcnt.
  __builtin_amdgcn_global_load_async_to_lds_b128(
      (__attribute__((address_space(1))) v4i*)(g),
      (__attribute__((address_space(3))) v4i*)(l), 0, 0);
#else
  *(uint4*)l = *(const uint4*)g;
#endif
}

__device__ __forceinline__ void async_wait_all() {
#if ASYNC_LDS
#if __has_builtin(__builtin_amdgcn_s_wait_asynccnt)
  __builtin_amdgcn_s_wait_asynccnt(0);
#else
  asm volatile("s_wait_asynccnt 0x0" ::: "memory");
#endif
#endif
}

__device__ __forceinline__ u16 f32_to_bf16(float f) {
  unsigned int u = __float_as_uint(f);
  u += 0x7FFFu + ((u >> 16) & 1u);   // round-to-nearest-even
  return (u16)(u >> 16);
}

__global__ void __launch_bounds__(256)
cvt_f32_bf16(const float* __restrict__ src, u16* __restrict__ dst, int n) {
  int i = blockIdx.x * blockDim.x + threadIdx.x;
  if (i < n) dst[i] = f32_to_bf16(src[i]);
}

// ---------------------------------------------------------------------------
// GEMM: Y[row, col] = sum_e X[row, e] * W[col, e] + bias[col]
//   X: [4096, 512] bf16 row-major, W: [512, 512] bf16 row-major (torch W layout)
//   8 waves / block; each wave: 16 rows x 64 cols (4 wmma accumulators)
//   MODE 0: store bf16 [B,H,S,DK]   (Q, K projections)
//   MODE 1: store bf16 [B,H,DK,S]   (V projection, transposed for PV gemm)
//   MODE 2: store f32  [B,S,E]      (output projection -> d_out)
// ---------------------------------------------------------------------------
template <int MODE>
__global__ void __launch_bounds__(256)
gemm_bf16(const u16* __restrict__ X, const u16* __restrict__ W,
          const float* __restrict__ bias, u16* __restrict__ outBf,
          float* __restrict__ outF) {
  const int lane  = threadIdx.x & 31;
  const int wave  = threadIdx.x >> 5;
  const int l16   = lane & 15;
  const int g     = lane >> 4;
  const int mBase = blockIdx.x * 128 + wave * 16;
  const int nBase = blockIdx.y * 64;

  f32x8 acc[4];
#pragma unroll
  for (int nt = 0; nt < 4; ++nt)
#pragma unroll
    for (int r = 0; r < 8; ++r) acc[nt][r] = 0.f;

  const u16* xRow = X + (mBase + l16) * EE;
  for (int kb = 0; kb < EE; kb += 32) {
    OpPack a;                               // A: M=l16, K = kb + (i/8)*16 + g*8 + i%8
    const u16* pa = xRow + kb + g * 8;
    a.q[0] = *(const uint4*)(pa);
    a.q[1] = *(const uint4*)(pa + 16);
#pragma unroll
    for (int nt = 0; nt < 4; ++nt) {        // B: N = col, K = kb + (i/8)*16 + g*8 + i%8
      OpPack bm;
      const u16* pb = W + (nBase + nt * 16 + l16) * EE + kb + g * 8;
      bm.q[0] = *(const uint4*)(pb);
      bm.q[1] = *(const uint4*)(pb + 16);
      acc[nt] = __builtin_amdgcn_wmma_f32_16x16x32_bf16(
          false, a.v, false, bm.v, (short)0, acc[nt], false, false);
    }
  }

#pragma unroll
  for (int nt = 0; nt < 4; ++nt) {
    const int col = nBase + nt * 16 + l16;
    const float bv = bias[col];
#pragma unroll
    for (int r = 0; r < 8; ++r) {
      const int row = mBase + r + 8 * g;     // D: M = r + 8g, N = l16
      const float val = acc[nt][r] + bv;
      if (MODE == 0) {
        const int bI = row >> 10, s = row & (SS - 1), h = col >> 6, d = col & (DKK - 1);
        outBf[((bI * HH + h) * SS + s) * DKK + d] = f32_to_bf16(val);
      } else if (MODE == 1) {
        const int bI = row >> 10, s = row & (SS - 1), h = col >> 6, d = col & (DKK - 1);
        outBf[((bI * HH + h) * DKK + d) * SS + s] = f32_to_bf16(val);
      } else {
        outF[row * EE + col] = val;
      }
    }
  }
}

// ---------------------------------------------------------------------------
// Fused flash attention per (b, h, 128 query rows).  8 waves x 16 q-rows.
// Inline per-element bias = cw2[h,:] . relu(dist*cw1 + cb1) + cb2[h].
// K/V tiles double-buffered in LDS, staged with async global->LDS DMA.
// ---------------------------------------------------------------------------
__global__ void __launch_bounds__(256)
attn_kernel(const u16* __restrict__ Qp, const u16* __restrict__ Kp,
            const u16* __restrict__ Vt, const float* __restrict__ dist,
            const int* __restrict__ mask,
            const float* __restrict__ cw1, const float* __restrict__ cb1,
            const float* __restrict__ cw2, const float* __restrict__ cb2,
            u16* __restrict__ attnX) {
  __shared__ u16 kTile[2][64 * 72];     // [j][d], padded stride 72
  __shared__ u16 vTile[2][64 * 72];     // [d][j], padded stride 72
  __shared__ u16 pBuf[8 * 16 * 72];     // per-wave P tile [M][K]

  const int tid  = threadIdx.x;
  const int lane = tid & 31;
  const int wave = tid >> 5;
  const int l16  = lane & 15;
  const int g    = lane >> 4;
  const int h    = blockIdx.y;
  const int b    = blockIdx.z;
  const int bh   = b * HH + h;
  const int qBase = blockIdx.x * 128 + wave * 16;

  float c1[HH], c1b[HH], w2[HH];
#pragma unroll
  for (int i = 0; i < HH; ++i) {
    c1[i]  = cw1[i];
    c1b[i] = cb1[i];
    w2[i]  = cw2[h * HH + i];
  }
  const float c2b = cb2[h];

  // Q A-operands for both K-steps of DK=64, loaded once
  OpPack aQ[2];
  {
    const u16* qrow = Qp + (bh * SS + qBase + l16) * DKK;
#pragma unroll
    for (int kb = 0; kb < 2; ++kb) {
      const u16* p = qrow + kb * 32 + g * 8;
      aQ[kb].q[0] = *(const uint4*)(p);
      aQ[kb].q[1] = *(const uint4*)(p + 16);
    }
  }

  float mRow[8], lRow[8];
#pragma unroll
  for (int r = 0; r < 8; ++r) { mRow[r] = -1e30f; lRow[r] = 0.f; }
  f32x8 oAcc[4];
#pragma unroll
  for (int nt = 0; nt < 4; ++nt)
#pragma unroll
    for (int r = 0; r < 8; ++r) oAcc[nt][r] = 0.f;

  // cooperative staging: 64 rows, 4 threads/row, 16 u16 (32B) per thread/tile
  const int rowIdx  = tid >> 2;
  const int quarter = tid & 3;
  auto stage = [&](int bufIdx, int j0) {
    const u16* srcK = Kp + (bh * SS + j0 + rowIdx) * DKK + quarter * 16;
    u16* dK = kTile[bufIdx] + rowIdx * 72 + quarter * 16;
    lds_copy16B(srcK,     dK);
    lds_copy16B(srcK + 8, dK + 8);
    const u16* srcV = Vt + (bh * DKK + rowIdx) * SS + j0 + quarter * 16;
    u16* dV = vTile[bufIdx] + rowIdx * 72 + quarter * 16;
    lds_copy16B(srcV,     dV);
    lds_copy16B(srcV + 8, dV + 8);
  };

  stage(0, 0);   // prologue: first tile in flight

  for (int it = 0; it < SS / 64; ++it) {
    const int j0  = it * 64;
    const int cur = it & 1;
    async_wait_all();      // this wave's staged tile has landed in LDS
    __syncthreads();       // all waves staged; prior reads of other buf done
    if (it + 1 < SS / 64) stage(cur ^ 1, j0 + 64);   // overlap next DMA

    const u16* kT = kTile[cur];
    const u16* vT = vTile[cur];

    // scores = Q . K^T   (contract over d)
    f32x8 st[4];
#pragma unroll
    for (int nt = 0; nt < 4; ++nt) {
#pragma unroll
      for (int r = 0; r < 8; ++r) st[nt][r] = 0.f;
#pragma unroll
      for (int kb = 0; kb < 2; ++kb) {
        OpPack bK;   // B: N = j (kTile row), K = d
        const u16* p = kT + (nt * 16 + l16) * 72 + kb * 32 + g * 8;
        bK.q[0] = *(const uint4*)(p);
        bK.q[1] = *(const uint4*)(p + 16);
        st[nt] = __builtin_amdgcn_wmma_f32_16x16x32_bf16(
            false, aQ[kb].v, false, bK.v, (short)0, st[nt], false, false);
      }
    }

    // scale, conv-bias, mask; batch the global loads per 16x16 tile so the
    // compiler can keep several loads in flight before the compute phase
#pragma unroll
    for (int nt = 0; nt < 4; ++nt) {
      float dv[8];
      int   mv[8];
#pragma unroll
      for (int r = 0; r < 8; ++r) {
        const int row = qBase + r + 8 * g;
        const int col = j0 + nt * 16 + l16;
        const int idx = (b * SS + row) * SS + col;
        dv[r] = dist[idx];
        mv[r] = mask[idx];
      }
#pragma unroll
      for (int r = 0; r < 8; ++r) {
        float biasV = c2b;
#pragma unroll
        for (int hh = 0; hh < HH; ++hh)
          biasV += w2[hh] * fmaxf(fmaf(dv[r], c1[hh], c1b[hh]), 0.f);
        const float sv = st[nt][r] * 0.125f * biasV;
        st[nt][r] = (mv[r] == 0) ? -1e9f : sv;
      }
    }

    // online softmax; row M = r + 8g lives in lanes sharing lane/16
#pragma unroll
    for (int r = 0; r < 8; ++r) {
      const float s0 = st[0][r], s1 = st[1][r], s2 = st[2][r], s3 = st[3][r];
      float tmax = fmaxf(fmaxf(s0, s1), fmaxf(s2, s3));
#pragma unroll
      for (int off = 1; off < 16; off <<= 1)
        tmax = fmaxf(tmax, __shfl_xor(tmax, off, 32));
      const float mNew  = fmaxf(mRow[r], tmax);
      const float scOld = __expf(mRow[r] - mNew);
      mRow[r] = mNew;
      const float p0 = __expf(s0 - mNew), p1 = __expf(s1 - mNew);
      const float p2 = __expf(s2 - mNew), p3 = __expf(s3 - mNew);
      float rsum = (p0 + p1) + (p2 + p3);
#pragma unroll
      for (int off = 1; off < 16; off <<= 1)
        rsum += __shfl_xor(rsum, off, 32);
      lRow[r] = lRow[r] * scOld + rsum;
      oAcc[0][r] *= scOld; oAcc[1][r] *= scOld;
      oAcc[2][r] *= scOld; oAcc[3][r] *= scOld;
      // D layout -> A layout relayering through per-wave LDS
      const int rowLoc = r + 8 * g;
      u16* pb = pBuf + wave * (16 * 72) + rowLoc * 72 + l16;
      pb[0]  = f32_to_bf16(p0);
      pb[16] = f32_to_bf16(p1);
      pb[32] = f32_to_bf16(p2);
      pb[48] = f32_to_bf16(p3);
    }

    // O += P . V  (contract over j); pBuf is per-wave, same-wave DS ordering
#pragma unroll
    for (int kb = 0; kb < 2; ++kb) {
      OpPack aP;   // A: M = l16, K = kb*32 + (i/8)*16 + g*8 + i%8
      const u16* p = pBuf + wave * (16 * 72) + l16 * 72 + kb * 32 + g * 8;
      aP.q[0] = *(const uint4*)(p);
      aP.q[1] = *(const uint4*)(p + 16);
#pragma unroll
      for (int nt = 0; nt < 4; ++nt) {
        OpPack bV;   // B: N = d (vTile row), K = j
        const u16* pv = vT + (nt * 16 + l16) * 72 + kb * 32 + g * 8;
        bV.q[0] = *(const uint4*)(pv);
        bV.q[1] = *(const uint4*)(pv + 16);
        oAcc[nt] = __builtin_amdgcn_wmma_f32_16x16x32_bf16(
            false, aP.v, false, bV.v, (short)0, oAcc[nt], false, false);
      }
    }
  }

  // finalize: divide by row sums, write bf16 [B,S,E] for the O projection
#pragma unroll
  for (int nt = 0; nt < 4; ++nt) {
#pragma unroll
    for (int r = 0; r < 8; ++r) {
      const int row = qBase + r + 8 * g;
      const int col = h * DKK + nt * 16 + l16;
      attnX[(b * SS + row) * EE + col] = f32_to_bf16(oAcc[nt][r] / lRow[r]);
    }
  }
}

extern "C" void kernel_launch(void* const* d_in, const int* in_sizes, int n_in,
                              void* d_out, int out_size, void* d_ws, size_t ws_size,
                              hipStream_t stream) {
  (void)in_sizes; (void)n_in; (void)out_size; (void)ws_size;
  const float* query = (const float*)d_in[0];
  const float* key   = (const float*)d_in[1];
  const float* value = (const float*)d_in[2];
  const float* dist  = (const float*)d_in[3];
  const int*   mask  = (const int*)d_in[4];
  const float* Wq = (const float*)d_in[5];  const float* bq = (const float*)d_in[6];
  const float* Wk = (const float*)d_in[7];  const float* bk = (const float*)d_in[8];
  const float* Wv = (const float*)d_in[9];  const float* bv = (const float*)d_in[10];
  const float* Wo = (const float*)d_in[11]; const float* bo = (const float*)d_in[12];
  const float* cw1 = (const float*)d_in[13]; const float* cb1 = (const float*)d_in[14];
  const float* cw2 = (const float*)d_in[15]; const float* cb2 = (const float*)d_in[16];

  u16* ws = (u16*)d_ws;
  const int WN = EE * EE;        // 262144 weight elements
  const int XN = BB * SS * EE;   // 2097152 activation elements
  u16* Wq_bf = ws;
  u16* Wk_bf = Wq_bf + WN;
  u16* Wv_bf = Wk_bf + WN;
  u16* Wo_bf = Wv_bf + WN;
  u16* Xq_bf = Wo_bf + WN;
  u16* Xk_bf = Xq_bf + XN;
  u16* Xv_bf = Xk_bf + XN;
  u16* Qp    = Xv_bf + XN;       // [B,H,S,DK]
  u16* Kp    = Qp + XN;          // [B,H,S,DK]
  u16* Vt    = Kp + XN;          // [B,H,DK,S]
  u16* attnX = Xq_bf;            // reuse: Xq consumed before attention runs

  cvt_f32_bf16<<<WN / 256, 256, 0, stream>>>(Wq, Wq_bf, WN);
  cvt_f32_bf16<<<WN / 256, 256, 0, stream>>>(Wk, Wk_bf, WN);
  cvt_f32_bf16<<<WN / 256, 256, 0, stream>>>(Wv, Wv_bf, WN);
  cvt_f32_bf16<<<WN / 256, 256, 0, stream>>>(Wo, Wo_bf, WN);
  cvt_f32_bf16<<<XN / 256, 256, 0, stream>>>(query, Xq_bf, XN);
  cvt_f32_bf16<<<XN / 256, 256, 0, stream>>>(key,   Xk_bf, XN);
  cvt_f32_bf16<<<XN / 256, 256, 0, stream>>>(value, Xv_bf, XN);

  dim3 gGemm(BB * SS / 128, EE / 64);
  gemm_bf16<0><<<gGemm, 256, 0, stream>>>(Xq_bf, Wq_bf, bq, Qp, nullptr);
  gemm_bf16<0><<<gGemm, 256, 0, stream>>>(Xk_bf, Wk_bf, bk, Kp, nullptr);
  gemm_bf16<1><<<gGemm, 256, 0, stream>>>(Xv_bf, Wv_bf, bv, Vt, nullptr);

  attn_kernel<<<dim3(SS / 128, HH, BB), 256, 0, stream>>>(
      Qp, Kp, Vt, dist, mask, cw1, cb1, cw2, cb2, attnX);

  gemm_bf16<2><<<gGemm, 256, 0, stream>>>(attnX, Wo_bf, bo, nullptr, (float*)d_out);
}
